// ShowAttendTellDecoder_68616397521315
// MI455X (gfx1250) — matrix operand order repaired and checked by reading, not compiled, not run on hardware
//
#include <hip/hip_runtime.h>
#include <hip/hip_bf16.h>

// ---------------------------------------------------------------------------
// Show-Attend-Tell decoder for MI455X (gfx1250, wave32, WMMA f16 16x16x32).
// All GEMMs run through v_wmma_f32_16x16x32_f16 with f32 accumulate.
// Each wave computes a 64x16 output strip (4 M-tiles): the large L2-resident
// B operand (weights) is read exactly once, and per K-chunk all fragment
// loads (1 B + 4 A) are issued as one batch before the 4 chained WMMAs so a
// single s_wait covers them and next-chunk loads can overlap WMMA execution.
// ---------------------------------------------------------------------------

typedef __attribute__((ext_vector_type(16))) _Float16 v16h;
typedef __attribute__((ext_vector_type(8)))  _Float16 v8h;
typedef __attribute__((ext_vector_type(8)))  float    v8f;

static constexpr int kB = 64;      // batch
static constexpr int kT = 100;     // time steps
static constexpr int kV = 10000;   // vocab
static constexpr int kE = 512;     // embed
static constexpr int kH = 512;     // hidden
static constexpr int kC = 1024;    // channels
static constexpr int kS = 256;     // spatial (16*16)
static constexpr int kA = 100;     // attention dim
static constexpr int kAP = 112;    // attention dim padded to 16

__device__ __forceinline__ float sigmoidf_(float x) { return 1.0f / (1.0f + expf(-x)); }

// ---------------------------------------------------------------------------
// 64x16 output strip: acc[i] = A[m0+16i : m0+16i+16, :K] * B[:K, n0:n0+16]
// A: f16 row-major (lda halves/row), B: f16 row-major (ldb halves/row).
// Fragment layouts per CDNA5 ISA 7.12.2 (16-bit A 16x32; B lane = N-column).
// ---------------------------------------------------------------------------
__device__ __forceinline__ void wmma_m4(const _Float16* __restrict__ A,
                                        const _Float16* __restrict__ B,
                                        int m0, int n0, int K,
                                        int lda, int ldb, v8f acc[4]) {
  const int lane = threadIdx.x & 31;
  const int row  = lane & 15;
  const int hi   = lane >> 4;              // 0: K 0-7/16-23, 1: K 8-15/24-31
  const _Float16* aBase = A + (size_t)(m0 + row) * lda + hi * 8;
  #pragma unroll 2
  for (int k0 = 0; k0 < K; k0 += 32) {
    // ---- batch all loads for this K-chunk ----
    // B fragment: lane L holds column n0+(L&15), K rows k0+8h+s / k0+16+8h+s
    const _Float16* bRow = B + (size_t)(k0 + hi * 8) * ldb + n0 + row;
    v8h blo, bhi;
    for (int s = 0; s < 8; ++s) { blo[s] = bRow[(size_t)s * ldb]; bhi[s] = bRow[(size_t)(s + 16) * ldb]; }
    v8h alo[4], ahi[4];
    #pragma unroll
    for (int i = 0; i < 4; ++i) {
      const _Float16* ap = aBase + (size_t)(i * 16) * lda + k0;
      alo[i] = *(const v8h*)(ap);
      ahi[i] = *(const v8h*)(ap + 16);
    }
    // ---- then the 4 WMMAs reusing the single B fragment ----
    v16h b = __builtin_shufflevector(blo, bhi, 0,1,2,3,4,5,6,7,8,9,10,11,12,13,14,15);
    #pragma unroll
    for (int i = 0; i < 4; ++i) {
      v16h a = __builtin_shufflevector(alo[i], ahi[i],
                                       0,1,2,3,4,5,6,7,8,9,10,11,12,13,14,15);
      acc[i] = __builtin_amdgcn_wmma_f32_16x16x32_f16(false, a, false, b,
                                                      (short)0, acc[i], false, false);
    }
  }
}

// ---------------------------------------------------------------------------
// Generic GEMM: C[m*ldc+n] = A*B + bias[n]   (f32 out). M % 64 == 0.
// ---------------------------------------------------------------------------
__global__ __launch_bounds__(256)
void gemm_bias_kernel(const _Float16* __restrict__ A, const _Float16* __restrict__ B,
                      const float* __restrict__ bias, float* __restrict__ C,
                      int M, int N, int K, int lda, int ldb, int ldc) {
  const int wave   = (int)((blockIdx.x * blockDim.x + threadIdx.x) >> 5);
  const int nwaves = (int)((gridDim.x * blockDim.x) >> 5);
  const int ntN = N >> 4;
  const int tasks = (M >> 6) * ntN;          // 64-row strips x 16-col tiles
  const int lane = threadIdx.x & 31;
  const int col = lane & 15, rb = (lane >> 4) * 8;
  for (int tIdx = wave; tIdx < tasks; tIdx += nwaves) {
    const int m0 = (tIdx / ntN) * 64;
    const int n0 = (tIdx % ntN) * 16;
    v8f acc[4] = {{}, {}, {}, {}};
    wmma_m4(A, B, m0, n0, K, lda, ldb, acc);
    const float bv = bias[n0 + col];
    #pragma unroll
    for (int i = 0; i < 4; ++i)
      #pragma unroll
      for (int r = 0; r < 8; ++r)
        C[(size_t)(m0 + i * 16 + rb + r) * ldc + n0 + col] = acc[i][r] + bv;
  }
}

// init = tanh(mean_feat @ feat_fc_W + b) -> h0 (n<512), c0 (n>=512). M=64.
__global__ __launch_bounds__(256)
void gemm_init_kernel(const _Float16* __restrict__ A, const _Float16* __restrict__ B,
                      const float* __restrict__ bias,
                      float* __restrict__ hbuf, float* __restrict__ cbuf) {
  const int wave   = (int)((blockIdx.x * blockDim.x + threadIdx.x) >> 5);
  const int nwaves = (int)((gridDim.x * blockDim.x) >> 5);
  const int tasks = (2 * kH) >> 4;
  const int lane = threadIdx.x & 31;
  const int col = lane & 15, rb = (lane >> 4) * 8;
  for (int tIdx = wave; tIdx < tasks; tIdx += nwaves) {
    const int n0 = tIdx * 16;
    v8f acc[4] = {{}, {}, {}, {}};
    wmma_m4(A, B, 0, n0, kC, kC, 2 * kH, acc);
    const int n = n0 + col;
    const float bv = bias[n];
    #pragma unroll
    for (int i = 0; i < 4; ++i)
      #pragma unroll
      for (int r = 0; r < 8; ++r) {
        const int m = i * 16 + rb + r;
        const float v = tanhf(acc[i][r] + bv);
        if (n < kH) hbuf[(size_t)m * kH + n] = v;
        else        cbuf[(size_t)m * kH + (n - kH)] = v;
      }
  }
}

// t16 = f16( tanh(xp @ Bpre + emb + (Lh_b + Lz_b)) ),  M=64, N=512, K=1536
__global__ __launch_bounds__(256)
void gemm_predpre_kernel(const _Float16* __restrict__ A, const _Float16* __restrict__ B,
                         const float* __restrict__ biaspre, const float* __restrict__ emb,
                         _Float16* __restrict__ t16) {
  const int wave   = (int)((blockIdx.x * blockDim.x + threadIdx.x) >> 5);
  const int nwaves = (int)((gridDim.x * blockDim.x) >> 5);
  const int tasks = kE >> 4;
  const int lane = threadIdx.x & 31;
  const int col = lane & 15, rb = (lane >> 4) * 8;
  for (int tIdx = wave; tIdx < tasks; tIdx += nwaves) {
    const int n0 = tIdx * 16;
    v8f acc[4] = {{}, {}, {}, {}};
    wmma_m4(A, B, 0, n0, kH + kC, kH + kC, kE, acc);
    const int n = n0 + col;
    const float bv = biaspre[n];
    #pragma unroll
    for (int i = 0; i < 4; ++i)
      #pragma unroll
      for (int r = 0; r < 8; ++r) {
        const int m = i * 16 + rb + r;
        t16[(size_t)m * kE + n] =
            (_Float16)tanhf(acc[i][r] + bv + emb[(size_t)m * kE + n]);
      }
  }
}

// ---------------------------------------------------------------------------
// Attention + softmax + context + emb/batchnorm + pack f16 LSTM inputs.
// One block per batch element (256 threads).
// ---------------------------------------------------------------------------
__global__ __launch_bounds__(256)
void attention_kernel(const float* __restrict__ hbuf, const float* __restrict__ attv,
                      const float* __restrict__ att_h_W, const float* __restrict__ att_out_W,
                      const float* __restrict__ att_out_b, const float* __restrict__ f_beta_W,
                      const float* __restrict__ f_beta_b, const _Float16* __restrict__ feat16,
                      const float* __restrict__ emb_table, const float* __restrict__ invstd,
                      const float* __restrict__ bn_mean, const float* __restrict__ bn_beta,
                      const int* __restrict__ reports, int t,
                      float* __restrict__ scores_out, float* __restrict__ emb_out,
                      float* __restrict__ gctx_out, _Float16* __restrict__ x16,
                      _Float16* __restrict__ xp16) {
  __shared__ float hloc[kH];
  __shared__ float hW[kAP];
  __shared__ float sc[kS];
  __shared__ float red[256];
  __shared__ float sh_beta, sh_max, sh_sum;
  const int b = blockIdx.x, tid = threadIdx.x;

  hloc[tid]       = hbuf[(size_t)b * kH + tid];
  hloc[tid + 256] = hbuf[(size_t)b * kH + 256 + tid];
  __syncthreads();

  // hW = h @ att_h_W  (512 x 100)
  if (tid < kA) {
    float s = 0.f;
    for (int k = 0; k < kH; ++k) s += hloc[k] * att_h_W[k * kA + tid];
    hW[tid] = s;
  }
  // beta = sigmoid(h @ f_beta_W + b)
  red[tid] = hloc[tid] * f_beta_W[tid] + hloc[tid + 256] * f_beta_W[tid + 256];
  __syncthreads();
  for (int off = 128; off > 0; off >>= 1) {
    if (tid < off) red[tid] += red[tid + off];
    __syncthreads();
  }
  if (tid == 0) sh_beta = sigmoidf_(red[0] + f_beta_b[0]);
  __syncthreads();

  // e_s = tanh(att_v + hW) . att_out_W + att_out_b   (s = tid)
  const float* av = attv + (size_t)(b * kS + tid) * kAP;
  float e = att_out_b[0];
  for (int j = 0; j < kA; ++j) e += tanhf(av[j] + hW[j]) * att_out_W[j];

  // softmax over 256 spatial positions
  red[tid] = e; __syncthreads();
  for (int off = 128; off > 0; off >>= 1) {
    if (tid < off) red[tid] = fmaxf(red[tid], red[tid + off]);
    __syncthreads();
  }
  if (tid == 0) sh_max = red[0];
  __syncthreads();
  const float ex = expf(e - sh_max);
  red[tid] = ex; __syncthreads();
  for (int off = 128; off > 0; off >>= 1) {
    if (tid < off) red[tid] += red[tid + off];
    __syncthreads();
  }
  if (tid == 0) sh_sum = red[0];
  __syncthreads();
  const float p = ex / sh_sum;
  sc[tid] = p;
  scores_out[(size_t)b * (kT * kS) + (size_t)t * kS + tid] = p;
  __syncthreads();

  const float beta = sh_beta;
  // ctx[c] = sum_s p[s] * feat[b,s,c]; g_ctx = beta * ctx
  for (int i = 0; i < 4; ++i) {
    const int ch = tid + 256 * i;
    float acc = 0.f;
    const _Float16* fb = feat16 + (size_t)b * kS * kC + ch;
    for (int s = 0; s < kS; ++s) acc += sc[s] * (float)fb[(size_t)s * kC];
    const float g = beta * acc;
    gctx_out[(size_t)b * kC + ch] = g;
    x16 [(size_t)b * 2048 + kE + ch] = (_Float16)g;
    xp16[(size_t)b * (kH + kC) + kH + ch] = (_Float16)g;
  }
  // emb (batchnormed lookup) + pack [emb | gctx | h] into x16
  const int w = (t == 0) ? 1 : reports[(size_t)b * kT + (t - 1)];
  for (int i = 0; i < 2; ++i) {
    const int j = tid + 256 * i;
    const float ev = (emb_table[(size_t)w * kE + j] - bn_mean[j]) * invstd[j] + bn_beta[j];
    emb_out[(size_t)b * kE + j] = ev;
    x16[(size_t)b * 2048 + j] = (_Float16)ev;
    x16[(size_t)b * 2048 + (kE + kC) + j] = (_Float16)hloc[j];
  }
}

// ---------------------------------------------------------------------------
// LSTM pointwise update. gates layout (B x 2048) = [i | f | g | o].
// ---------------------------------------------------------------------------
__global__ __launch_bounds__(512)
void lstm_kernel(const float* __restrict__ gates, float* __restrict__ hbuf,
                 float* __restrict__ cbuf, _Float16* __restrict__ xp16) {
  const int b = blockIdx.x, n = threadIdx.x;
  const float* g = gates + (size_t)b * (4 * kH);
  const float ig = sigmoidf_(g[n]);
  const float fg = sigmoidf_(g[n + kH]);
  const float gg = tanhf(g[n + 2 * kH]);
  const float og = sigmoidf_(g[n + 3 * kH]);
  const float cn = fg * cbuf[(size_t)b * kH + n] + ig * gg;
  const float hn = og * tanhf(cn);
  cbuf[(size_t)b * kH + n] = cn;
  hbuf[(size_t)b * kH + n] = hn;
  xp16[(size_t)b * (kH + kC) + n] = (_Float16)hn;
}

// ---------------------------------------------------------------------------
// Prologue kernels
// ---------------------------------------------------------------------------
__global__ void feat_transpose_kernel(const float* __restrict__ img, _Float16* __restrict__ feat16) {
  const size_t id = (size_t)blockIdx.x * blockDim.x + threadIdx.x;   // (b*S+s)*C + c
  if (id >= (size_t)kB * kS * kC) return;
  const int c = (int)(id & (kC - 1));
  const int s = (int)((id >> 10) & (kS - 1));
  const int b = (int)(id >> 18);
  feat16[id] = (_Float16)img[((size_t)b * kC + c) * kS + s];
}

__global__ __launch_bounds__(256)
void mean_kernel(const float* __restrict__ img, _Float16* __restrict__ mean16) {
  const int b = blockIdx.x, tid = threadIdx.x;
  for (int i = 0; i < 4; ++i) {
    const int ch = tid + 256 * i;
    const float* p = img + ((size_t)b * kC + ch) * kS;
    float s = 0.f;
    for (int j = 0; j < kS; ++j) s += p[j];
    mean16[(size_t)b * kC + ch] = (_Float16)(s * (1.0f / (float)kS));
  }
}

__global__ void cvt_f16_kernel(const float* __restrict__ in, _Float16* __restrict__ out, int n) {
  const int id = blockIdx.x * blockDim.x + threadIdx.x;
  if (id < n) out[id] = (_Float16)in[id];
}

// Wcomb[k][n], K=2048 (emb|gctx|h), N=2048 gates: transpose of [W_ih | W_hh]
__global__ void build_wcomb_kernel(const float* __restrict__ W_ih, const float* __restrict__ W_hh,
                                   _Float16* __restrict__ out) {
  const int id = blockIdx.x * blockDim.x + threadIdx.x;
  if (id >= 2048 * 2048) return;
  const int k = id / 2048, n = id % 2048;
  const float v = (k < kE + kC) ? W_ih[(size_t)n * (kE + kC) + k]
                                : W_hh[(size_t)n * kH + (k - (kE + kC))];
  out[id] = (_Float16)v;
}

// Bpre[k][n], K=1536 (h_new|gctx), N=512: [Lh_W ; Lz_W]
__global__ void build_bpre_kernel(const float* __restrict__ Lh_W, const float* __restrict__ Lz_W,
                                  _Float16* __restrict__ out) {
  const int id = blockIdx.x * blockDim.x + threadIdx.x;
  if (id >= (kH + kC) * kE) return;
  const int k = id / kE, n = id % kE;
  const float v = (k < kH) ? Lh_W[(size_t)k * kE + n] : Lz_W[(size_t)(k - kH) * kE + n];
  out[id] = (_Float16)v;
}

// att_v_W padded (1024 x 112), zero-filled beyond n=100
__global__ void build_attvw_kernel(const float* __restrict__ att_v_W, _Float16* __restrict__ out) {
  const int id = blockIdx.x * blockDim.x + threadIdx.x;
  if (id >= kC * kAP) return;
  const int k = id / kAP, n = id % kAP;
  out[id] = (_Float16)((n < kA) ? att_v_W[(size_t)k * kA + n] : 0.0f);
}

__global__ void small_params_kernel(const float* __restrict__ b_ih, const float* __restrict__ b_hh,
                                    const float* __restrict__ Lh_b, const float* __restrict__ Lz_b,
                                    const float* __restrict__ bn_gamma, const float* __restrict__ bn_var,
                                    const float* __restrict__ att_v_b,
                                    float* __restrict__ gatebias, float* __restrict__ biaspre,
                                    float* __restrict__ invstd, float* __restrict__ attvb) {
  const int id = blockIdx.x * blockDim.x + threadIdx.x;
  if (id < 4 * kH) gatebias[id] = b_ih[id] + b_hh[id];
  if (id < kE) {
    biaspre[id] = Lh_b[id] + Lz_b[id];
    invstd[id] = rsqrtf(bn_var[id] + 1e-5f) * bn_gamma[id];
  }
  if (id < kAP) attvb[id] = (id < kA) ? att_v_b[id] : 0.0f;
}

// ---------------------------------------------------------------------------
extern "C" void kernel_launch(void* const* d_in, const int* in_sizes, int n_in,
                              void* d_out, int out_size, void* d_ws, size_t ws_size,
                              hipStream_t stream) {
  const float* image_features = (const float*)d_in[0];
  const int*   reports        = (const int*)d_in[1];
  const float* feat_fc_W = (const float*)d_in[2];
  const float* feat_fc_b = (const float*)d_in[3];
  const float* f_beta_W  = (const float*)d_in[4];
  const float* f_beta_b  = (const float*)d_in[5];
  const float* emb_table = (const float*)d_in[6];
  const float* bn_gamma  = (const float*)d_in[7];
  const float* bn_beta   = (const float*)d_in[8];
  const float* bn_mean   = (const float*)d_in[9];
  const float* bn_var    = (const float*)d_in[10];
  const float* att_v_W   = (const float*)d_in[11];
  const float* att_v_b   = (const float*)d_in[12];
  const float* att_h_W   = (const float*)d_in[13];
  const float* att_out_W = (const float*)d_in[14];
  const float* att_out_b = (const float*)d_in[15];
  const float* W_ih = (const float*)d_in[16];
  const float* W_hh = (const float*)d_in[17];
  const float* b_ih = (const float*)d_in[18];
  const float* b_hh = (const float*)d_in[19];
  const float* Lh_W = (const float*)d_in[20];
  const float* Lh_b = (const float*)d_in[21];
  const float* Lz_W = (const float*)d_in[22];
  const float* Lz_b = (const float*)d_in[23];
  const float* Lo_W = (const float*)d_in[24];
  const float* Lo_b = (const float*)d_in[25];

  float* preds      = (float*)d_out;                              // (B,T,V)
  float* scores_out = (float*)d_out + (size_t)kB * kT * kV;       // (B,T,S)

  // ---- workspace carve (256B aligned) ----
  char* wp = (char*)d_ws;
  auto alloc = [&](size_t bytes) -> void* {
    void* p = (void*)wp;
    wp += (bytes + 255) & ~(size_t)255;
    return p;
  };
  _Float16* feat16   = (_Float16*)alloc((size_t)kB * kS * kC * 2);      // 32 MB
  _Float16* Lo16     = (_Float16*)alloc((size_t)kE * kV * 2);           // 10 MB
  _Float16* Wcomb    = (_Float16*)alloc((size_t)2048 * 2048 * 2);       // 8 MB
  _Float16* featfc16 = (_Float16*)alloc((size_t)kC * 2 * kH * 2);       // 2 MB
  _Float16* Bpre     = (_Float16*)alloc((size_t)(kH + kC) * kE * 2);
  _Float16* attvW16  = (_Float16*)alloc((size_t)kC * kAP * 2);
  _Float16* mean16   = (_Float16*)alloc((size_t)kB * kC * 2);
  _Float16* x16      = (_Float16*)alloc((size_t)kB * 2048 * 2);
  _Float16* xp16     = (_Float16*)alloc((size_t)kB * (kH + kC) * 2);
  _Float16* t16      = (_Float16*)alloc((size_t)kB * kE * 2);
  float* attv     = (float*)alloc((size_t)kB * kS * kAP * 4);           // 7.3 MB
  float* gates    = (float*)alloc((size_t)kB * 4 * kH * 4);
  float* hbuf     = (float*)alloc((size_t)kB * kH * 4);
  float* cbuf     = (float*)alloc((size_t)kB * kH * 4);
  float* embb     = (float*)alloc((size_t)kB * kE * 4);
  float* gctx     = (float*)alloc((size_t)kB * kC * 4);
  float* gatebias = (float*)alloc((size_t)4 * kH * 4);
  float* biaspre  = (float*)alloc((size_t)kE * 4);
  float* invstd   = (float*)alloc((size_t)kE * 4);
  float* attvb    = (float*)alloc((size_t)kAP * 4);

  // ---- prologue: repack / convert ----
  feat_transpose_kernel<<<(kB * kS * kC) / 256, 256, 0, stream>>>(image_features, feat16);
  mean_kernel<<<kB, 256, 0, stream>>>(image_features, mean16);
  cvt_f16_kernel<<<(kE * kV + 255) / 256, 256, 0, stream>>>(Lo_W, Lo16, kE * kV);
  cvt_f16_kernel<<<(kC * 2 * kH + 255) / 256, 256, 0, stream>>>(feat_fc_W, featfc16, kC * 2 * kH);
  build_wcomb_kernel<<<(2048 * 2048) / 256, 256, 0, stream>>>(W_ih, W_hh, Wcomb);
  build_bpre_kernel<<<((kH + kC) * kE) / 256, 256, 0, stream>>>(Lh_W, Lz_W, Bpre);
  build_attvw_kernel<<<(kC * kAP + 255) / 256, 256, 0, stream>>>(att_v_W, attvW16);
  small_params_kernel<<<(4 * kH) / 256, 256, 0, stream>>>(b_ih, b_hh, Lh_b, Lz_b, bn_gamma,
                                                          bn_var, att_v_b, gatebias, biaspre,
                                                          invstd, attvb);
  // h0, c0 = tanh(mean_feat @ feat_fc_W + b)   (64 x 1024, K=1024): 64 tasks
  gemm_init_kernel<<<8, 256, 0, stream>>>(mean16, featfc16, feat_fc_b, hbuf, cbuf);
  // att_v = feat_flat @ att_v_W + b  (16384 x 112, K=1024): 256*7 = 1792 tasks
  gemm_bias_kernel<<<224, 256, 0, stream>>>(feat16, attvW16, attvb, attv,
                                            kB * kS, kAP, kC, kC, kAP, kAP);

  // ---- recurrence ----
  for (int t = 0; t < kT; ++t) {
    attention_kernel<<<kB, 256, 0, stream>>>(hbuf, attv, att_h_W, att_out_W, att_out_b,
                                             f_beta_W, f_beta_b, feat16, emb_table, invstd,
                                             bn_mean, bn_beta, reports, t,
                                             scores_out, embb, gctx, x16, xp16);
    // gates = [emb|gctx|h] @ Wcomb + (b_ih + b_hh)   (64 x 2048, K=2048): 128 tasks
    gemm_bias_kernel<<<16, 256, 0, stream>>>(x16, Wcomb, gatebias, gates,
                                             kB, 4 * kH, 2048, 2048, 4 * kH, 4 * kH);
    lstm_kernel<<<kB, 512, 0, stream>>>(gates, hbuf, cbuf, xp16);
    // t16 = tanh([h_new|gctx] @ [Lh_W;Lz_W] + emb + Lh_b + Lz_b)  (64x512,K=1536): 32 tasks
    gemm_predpre_kernel<<<4, 256, 0, stream>>>(xp16, Bpre, biaspre, embb, t16);
    // preds[:,t,:] = t16 @ Lo_W + Lo_b   (64 x 10000, K=512): 625 tasks
    gemm_bias_kernel<<<79, 256, 0, stream>>>(t16, Lo16, Lo_b, preds + (size_t)t * kV,
                                             kB, kV, kE, kE, kV, kT * kV);
  }
}
